// Contrast_Loss_sig_773094114106
// MI455X (gfx1250) — matrix operand order repaired and checked by reading, not compiled, not run on hardware
//
#include <hip/hip_runtime.h>
#include <hip/hip_bf16.h>
#include <stdint.h>

// ---------------- problem constants (from reference) ----------------
constexpr int kB    = 4;
constexpr int kC    = 256;
constexpr int kHW   = 128 * 128;     // 16384
constexpr int kS    = 19;
constexpr int kN    = kB * kHW;      // 65536
constexpr int kQ    = 256;
constexpr int kNeg  = 512;
constexpr int kPx   = 32;            // pixels per tile/chunk
constexpr int kChunks = kN / kPx;    // 2048

constexpr float kTemp   = 0.5f;
constexpr float kThresh = 0.97f;
constexpr float kAlpha  = 0.99f;
constexpr float kEps    = 1e-8f;

typedef __attribute__((ext_vector_type(16))) _Float16 v16h;
typedef __attribute__((ext_vector_type(8)))  float    v8f;
typedef __attribute__((ext_vector_type(4)))  unsigned v4u;
typedef __attribute__((ext_vector_type(4)))  int      v4i;
typedef __attribute__((ext_vector_type(8)))  int      v8i;

// ---------------- threefry2x32 (JAX-style counter RNG) ----------------
__device__ __forceinline__ uint32_t rotl32(uint32_t x, uint32_t n) {
  return (x << n) | (x >> (32u - n));
}
__device__ __forceinline__ void threefry2x32(uint32_t k0, uint32_t k1,
                                             uint32_t c0, uint32_t c1,
                                             uint32_t& o0, uint32_t& o1) {
  const uint32_t rot[8] = {13u, 15u, 26u, 6u, 17u, 29u, 16u, 24u};
  uint32_t ks[3] = {k0, k1, 0x1BD11BDAu ^ k0 ^ k1};
  uint32_t x0 = c0 + ks[0];
  uint32_t x1 = c1 + ks[1];
#pragma unroll
  for (int i = 0; i < 5; ++i) {
#pragma unroll
    for (int r = 0; r < 4; ++r) {
      x0 += x1;
      x1 = rotl32(x1, rot[(i & 1) * 4 + r]);
      x1 ^= x0;
    }
    x0 += ks[(i + 1) % 3];
    x1 += ks[(i + 2) % 3] + (uint32_t)(i + 1);
  }
  o0 = x0; o1 = x1;
}
__device__ __forceinline__ float rand_u01(uint32_t k0, uint32_t k1,
                                          uint32_t c0, uint32_t c1) {
  uint32_t a, b;
  threefry2x32(k0, k1, c0, c1, a, b);
  return (float)(a >> 8) * (1.0f / 16777216.0f);
}

// ---------------- K0: zero accumulators (must run every launch) ----------------
__global__ void k0_zero(float* protoSum, float* lossAcc) {
  int t = blockIdx.x * 256 + threadIdx.x;
  if (t < kS * kC) protoSum[t] = 0.0f;
  if (t == 0) lossAcc[0] = 0.0f;
}

// ---------------- K1: fused transpose + class/hard + norms + proto scatter ----------------
__global__ void __launch_bounds__(256)
k1_fused(const float* __restrict__ rep, const float* __restrict__ label,
         const float* __restrict__ mask, const float* __restrict__ prob,
         float* __restrict__ repT, int* __restrict__ clsArr,
         float* __restrict__ pixNorm, float* __restrict__ protoSum,
         unsigned* __restrict__ validCntChunk, unsigned* __restrict__ hardCntChunk) {
  __shared__ float tile[kPx][kC + 4];          // 32 x 260 f32
  __shared__ float protoAcc[kS * kC];          // 19 x 256 f32
  __shared__ int clsL[kPx];
  __shared__ float normP[kPx][8];
  __shared__ unsigned cntV[kS], cntH[kS];

  const int t = threadIdx.x;
  const int blk = blockIdx.x;
  const int n0 = blk * kPx;
  const int b = n0 >> 14;
  const int hw0 = n0 & (kHW - 1);

  for (int i = t; i < kS * kC; i += 256) protoAcc[i] = 0.0f;
  if (t < kS) { cntV[t] = 0u; cntH[t] = 0u; }

  const int w = t >> 5, lane = t & 31;
#pragma unroll 4
  for (int it = 0; it < 32; ++it) {
    int c = it * 8 + w;
    tile[lane][c] = rep[((size_t)(b * kC + c) << 14) + hw0 + lane];
  }
  __syncthreads();

  if (t < kPx) {
    int hw = hw0 + t;
    float mk = mask[((size_t)b << 14) + hw];
    int cls = -1;
    for (int s = 0; s < kS; ++s) {
      float lv = label[((size_t)(b * kS + s) << 14) + hw];
      if (lv > 0.5f && mk > 0.0f) cls = s;
    }
    int hard = 0;
    if (cls >= 0) {
      float pv = prob[((size_t)(b * kS + cls) << 14) + hw];
      hard = (pv < kThresh) ? 1 : 0;
      atomicAdd(&cntV[cls], 1u);
      if (hard) atomicAdd(&cntH[cls], 1u);
    }
    clsL[t] = cls;
    clsArr[n0 + t] = (cls < 0) ? -1 : (cls | (hard << 8));
  }

  {
    int p = t >> 3, part = t & 7;
    float acc = 0.0f;
#pragma unroll
    for (int cc = 0; cc < 32; ++cc) {
      float v = tile[p][part * 32 + cc];
      acc += v * v;
    }
    normP[p][part] = acc;
  }
  __syncthreads();

  if (t < kPx) {
    float s2 = 0.0f;
#pragma unroll
    for (int k = 0; k < 8; ++k) s2 += normP[t][k];
    pixNorm[n0 + t] = sqrtf(s2);
  }

  for (int p = 0; p < kPx; ++p) {
    float v = tile[p][t];
    int s = clsL[p];
    if (s >= 0) atomicAdd(&protoAcc[s * kC + t], v);
    repT[(size_t)(n0 + p) * kC + t] = v;
  }
  __syncthreads();

  for (int i = t; i < kS * kC; i += 256) {
    float v = protoAcc[i];
    if (v != 0.0f) atomicAdd(&protoSum[i], v);
  }
  if (t < kS) {
    validCntChunk[t * kChunks + blk] = cntV[t];
    hardCntChunk[t * kChunks + blk] = cntH[t];
  }
}

// ---------------- K2: per-segment scan of chunk counts ----------------
__global__ void k2_scan(const unsigned* __restrict__ validCntChunk,
                        const unsigned* __restrict__ hardCntChunk,
                        unsigned* __restrict__ chunkBaseValid,
                        unsigned* __restrict__ chunkBaseHard,
                        unsigned* __restrict__ validTot, unsigned* __restrict__ hardTot,
                        unsigned* __restrict__ validSegBase, unsigned* __restrict__ hardSegBase,
                        unsigned* __restrict__ hardOk) {
  int t = threadIdx.x;
  if (t < kS) {
    unsigned run = 0;
    for (int c = 0; c < kChunks; ++c) {
      chunkBaseValid[t * kChunks + c] = run;
      run += validCntChunk[t * kChunks + c];
    }
    validTot[t] = run;
  } else if (t >= 32 && t < 32 + kS) {
    int s = t - 32;
    unsigned run = 0;
    for (int c = 0; c < kChunks; ++c) {
      chunkBaseHard[s * kChunks + c] = run;
      run += hardCntChunk[s * kChunks + c];
    }
    hardTot[s] = run;
    hardOk[s] = (run > 0u) ? 1u : 0u;
  }
  __syncthreads();
  if (t == 0) {
    unsigned rv = 0, rh = 0;
    for (int s = 0; s < kS; ++s) {
      validSegBase[s] = rv; rv += validTot[s];
      hardSegBase[s]  = rh; rh += hardTot[s];
    }
  }
}

// ---------------- K3: order-preserving compaction (1 wave32 per chunk) ----------------
__global__ void __launch_bounds__(32)
k3_compact(const int* __restrict__ clsArr,
           const unsigned* __restrict__ chunkBaseValid,
           const unsigned* __restrict__ chunkBaseHard,
           const unsigned* __restrict__ validSegBase,
           const unsigned* __restrict__ hardSegBase,
           int* __restrict__ validList, int* __restrict__ hardList) {
  int lane = threadIdx.x;
  int blk = blockIdx.x;
  int n = blk * kPx + lane;
  int cv = clsArr[n];
  int sMine = (cv < 0) ? -1 : (cv & 0xff);
  int hard = (cv >= 0) && (cv & 0x100);
  unsigned ltMask = (lane == 31) ? 0x7fffffffu : ((1u << lane) - 1u);
  for (int s = 0; s < kS; ++s) {
    unsigned mv = (unsigned)__ballot(sMine == s);
    if (sMine == s) {
      unsigned pos = (unsigned)__popc(mv & ltMask);
      validList[validSegBase[s] + chunkBaseValid[s * kChunks + blk] + pos] = n;
    }
    unsigned mh = (unsigned)__ballot((sMine == s) && hard);
    if ((sMine == s) && hard) {
      unsigned pos = (unsigned)__popc(mh & ltMask);
      hardList[hardSegBase[s] + chunkBaseHard[s * kChunks + blk] + pos] = n;
    }
  }
}

// ---------------- K4: prototype finalize + WMMA Gram matrix + class CDF ----------------
__global__ void __launch_bounds__(256)
k4_proto(const float* __restrict__ protosIn, const float* __restrict__ protoSum,
         const unsigned* __restrict__ validTot,
         float* __restrict__ proto, float* __restrict__ protoNorm,
         float* __restrict__ simCdf) {
  __shared__ _Float16 ph[32][kC];
  __shared__ float sim[32][32];
  __shared__ float isNew[kS];
  __shared__ float pn[kS];
  int t = threadIdx.x;

  if (t < kS) {
    float rs = 0.0f;
    for (int c = 0; c < kC; ++c) rs += protosIn[t * kC + c];
    isNew[t] = (rs == 0.0f) ? 1.0f : 0.0f;
  }
  __syncthreads();

  for (int s = 0; s < kS; ++s) {
    float mean = protoSum[s * kC + t] / fmaxf((float)validTot[s], 1.0f);
    float v = (isNew[s] != 0.0f) ? mean
                                 : (kAlpha * protosIn[s * kC + t] + (1.0f - kAlpha) * mean);
    proto[s * kC + t] = v;
    ph[s][t] = (_Float16)v;
  }
  for (int s = kS; s < 32; ++s) ph[s][t] = (_Float16)0.0f;
  __syncthreads();

  if (t < kS) {
    float s2 = 0.0f;
    for (int c = 0; c < kC; ++c) { float v = proto[t * kC + c]; s2 += v * v; }
    pn[t] = sqrtf(s2);
    protoNorm[t] = pn[t];
  }
  __syncthreads();

  // Gram = P * P^T via v_wmma_f32_16x16x32_f16 (2x2 tiles, K=256 in 8 steps)
  if (t < 32) {
    int lane = t, half = lane >> 4, l16 = lane & 15;
    for (int mi = 0; mi < 2; ++mi) {
      for (int ni = 0; ni < 2; ++ni) {
        v8f acc = {};
        int row = mi * 16 + l16;
        int col = ni * 16 + l16;
        for (int k0 = 0; k0 < kC; k0 += 32) {
          v16h a, bm;
#pragma unroll
          for (int e = 0; e < 8; ++e) {
            a[e]     = ph[row][k0 + half * 8 + e];
            a[8 + e] = ph[row][k0 + 16 + half * 8 + e];
          }
#pragma unroll
          for (int e = 0; e < 8; ++e) {
            bm[e]     = ph[col][k0 + half * 8 + e];
            bm[8 + e] = ph[col][k0 + 16 + half * 8 + e];
          }
          acc = __builtin_amdgcn_wmma_f32_16x16x32_f16(
              false, a, false, bm, (short)0, acc, false, false);
        }
#pragma unroll
        for (int r = 0; r < 8; ++r)
          sim[mi * 16 + r + half * 8][ni * 16 + l16] = acc[r];
      }
    }
  }
  __syncthreads();

  if (t < kS) {
    int i = t;
    float sims[kS - 1];
    float mx = -1e30f;
    for (int tt = 0; tt < kS - 1; ++tt) {
      int j = (i + 1 + tt) % kS;
      float cs = sim[i][j] / fmaxf(pn[i] * pn[j], kEps);
      sims[tt] = cs / kTemp;
      mx = fmaxf(mx, sims[tt]);
    }
    float z = 0.0f;
    for (int tt = 0; tt < kS - 1; ++tt) { sims[tt] = expf(sims[tt] - mx); z += sims[tt]; }
    float run = 0.0f;
    for (int tt = 0; tt < kS - 1; ++tt) {
      run += sims[tt] / z;
      simCdf[i * (kS - 1) + tt] = run;
    }
  }
}

// ---------------- K5a: anchor sampling ----------------
__global__ void k5_anchors(const int* __restrict__ hardList,
                           const unsigned* __restrict__ hardSegBase,
                           const unsigned* __restrict__ hardTot,
                           int* __restrict__ anchorIdx) {
  int s = blockIdx.x;
  int q = threadIdx.x;
  unsigned tot = hardTot[s];
  float u = rand_u01(0xA17C5u, (uint32_t)s, (uint32_t)q, 0x1u);
  unsigned k = (tot > 0u) ? min((unsigned)(u * (float)tot), tot - 1u) : 0u;
  anchorIdx[s * kQ + q] = (tot > 0u) ? hardList[hardSegBase[s] + k] : 0;
}

// ---------------- K5b: negative-pool sampling ----------------
__global__ void k5_pool(const int* __restrict__ validList,
                        const unsigned* __restrict__ validSegBase,
                        const unsigned* __restrict__ validTot,
                        int* __restrict__ poolIdx) {
  int gid = blockIdx.x * 256 + threadIdx.x;
  if (gid >= kS * kQ * kNeg) return;
  int s = gid / (kQ * kNeg);
  int slot = gid % (kQ * kNeg);
  unsigned tot = validTot[s];
  float u = rand_u01(0x9001u, (uint32_t)s, (uint32_t)slot, 0x2u);
  unsigned k = (tot > 0u) ? min((unsigned)(u * (float)tot), tot - 1u) : 0u;
  poolIdx[gid] = (tot > 0u) ? validList[validSegBase[s] + k] : 0;
}

// ---------------- K6: main contrastive loss ----------------
// One block per (anchor q, segment s). 8 waves x 64 negatives.
// Anchor staged by TDM tensor_load_to_lds; negatives double-buffered through LDS
// with GLOBAL_LOAD_ASYNC_TO_LDS_B128 (ASYNCcnt pipelining).
__global__ void __launch_bounds__(256)
k6_loss(const float* __restrict__ repT, const float* __restrict__ pixNorm,
        const float* __restrict__ proto, const float* __restrict__ protoNorm,
        const int* __restrict__ anchorIdx, const int* __restrict__ poolIdx,
        const float* __restrict__ simCdf, const unsigned* __restrict__ hardOk,
        float* __restrict__ lossAcc) {
  __shared__ float anc[kC];
  __shared__ float logits[1 + kNeg];
  __shared__ float red[256];
  __shared__ float nbuf[8][2][kC];   // per-wave double buffer for negatives (16 KB)

  const int q = blockIdx.x;
  const int s = blockIdx.y;
  const int t = threadIdx.x;
  const int lane = t & 31, w = t >> 5;

  const int aIdx = anchorIdx[s * kQ + q];

#if __has_builtin(__builtin_amdgcn_tensor_load_to_lds) && __has_builtin(__builtin_amdgcn_s_wait_tensorcnt)
  // TDM: DMA the 1KB anchor row into LDS; only wave 0 issues & waits (TENSORcnt is per-wave).
  if (t < 32) {
    uint64_t ga = (uint64_t)(uintptr_t)(repT + (size_t)aIdx * kC);
    uint32_t lds = (uint32_t)(uintptr_t)(void*)anc;  // ISA: LDS aperture keeps offset in addr[31:0]
    v4u g0;
    g0.x = 1u;                                        // count=1 (valid user descriptor)
    g0.y = lds;                                       // lds_addr
    g0.z = (uint32_t)ga;                              // global_addr[31:0]
    g0.w = (uint32_t)((ga >> 32) & 0x01FFFFFFu) | 0x80000000u;  // addr[56:32] | type=2
    v8i g1;
    g1[0] = 0x00020000;                               // data_size = 4 bytes
    g1[1] = (int)(256u << 16);                        // tensor_dim0 = 256 (bits 79:48, low 16)
    g1[2] = (int)(1u << 16);                          // tensor_dim1 = 1
    g1[3] = (int)(256u << 16);                        // tile_dim0 = 256
    g1[4] = 1;                                        // tile_dim1 = 1, tile_dim2 = 0
    g1[5] = 256;                                      // tensor_dim0_stride = 256
    g1[6] = (int)(256u << 16);                        // tensor_dim1_stride (low 16)
    g1[7] = 0;
    v4i gz = {0, 0, 0, 0};
#if __clang_major__ >= 23
    v8i gz8 = {0, 0, 0, 0, 0, 0, 0, 0};
    __builtin_amdgcn_tensor_load_to_lds(g0, g1, gz, gz, gz8, 0);
#else
    __builtin_amdgcn_tensor_load_to_lds(g0, g1, gz, gz, 0);
#endif
    __builtin_amdgcn_s_wait_tensorcnt((short)0);
  }
#else
  anc[t] = repT[(size_t)aIdx * kC + t];
#endif
  __syncthreads();

  // positive logit: cosine(anchor, proto[s]) / TEMP
  red[t] = anc[t] * proto[s * kC + t];
  __syncthreads();
  for (int off = 128; off > 0; off >>= 1) {
    if (t < off) red[t] += red[t + off];
    __syncthreads();
  }
  const float aNorm = pixNorm[aIdx];
  if (t == 0)
    logits[0] = (red[0] / fmaxf(aNorm * protoNorm[s], kEps)) / kTemp;

  float a0[8];
#pragma unroll
  for (int e = 0; e < 8; ++e) a0[e] = anc[lane * 8 + e];

  const uint64_t repTbase = (uint64_t)(uintptr_t)repT;
  // async-stage one 1KB negative vector into this wave's LDS buffer:
  // two b128 copies per lane (INST_OFFSET applies to both LDS and global address).
  auto stage = [&](int nIdx, int buf) {
    uint32_t dst = (uint32_t)(uintptr_t)&nbuf[w][buf][lane * 8];
    int goff = (int)(((size_t)nIdx * kC + (size_t)lane * 8) * sizeof(float));
    asm volatile("global_load_async_to_lds_b128 %0, %1, %2\n\t"
                 "global_load_async_to_lds_b128 %0, %1, %2 offset:16"
                 :: "v"(dst), "v"(goff), "s"(repTbase) : "memory");
  };
  auto sampleNegIdx = [&](int j) -> int {
    float u = rand_u01(0xCA7CA7u, (uint32_t)s, (uint32_t)q, (uint32_t)j);
    int samp = kS - 2;
    for (int tt = 0; tt < kS - 1; ++tt) {
      if (u <= simCdf[s * (kS - 1) + tt]) { samp = tt; break; }
    }
    int negS = (s + 1 + samp) % kS;
    return poolIdx[negS * (kQ * kNeg) + q * kNeg + j];
  };

  const int jBase = w * (kNeg / 8);
  int nIdxCur = sampleNegIdx(jBase);
  stage(nIdxCur, 0);

  for (int jj = 0; jj < kNeg / 8; ++jj) {
    const int cur = jj & 1;
    int nIdxNext = 0;
    if (jj + 1 < kNeg / 8) {
      nIdxNext = sampleNegIdx(jBase + jj + 1);
      __builtin_prefetch(pixNorm + nIdxNext, 0, 0);   // global_prefetch_b8
      stage(nIdxNext, 1 - cur);
      asm volatile("s_wait_asynccnt 2" ::: "memory"); // current buffer complete
    } else {
      asm volatile("s_wait_asynccnt 0" ::: "memory");
    }
    const float* nv = &nbuf[w][cur][lane * 8];
    float d = 0.0f;
#pragma unroll
    for (int e = 0; e < 8; ++e) d += a0[e] * nv[e];
#pragma unroll
    for (int off = 16; off > 0; off >>= 1) d += __shfl_xor(d, off, 32);
    if (lane == 0) {
      float nN = pixNorm[nIdxCur];
      logits[1 + jBase + jj] = (d / fmaxf(aNorm * nN, kEps)) / kTemp;
    }
    nIdxCur = nIdxNext;
  }
  __syncthreads();

  // block logsumexp over 513 logits -> CE vs slot 0
  float mx = -1e30f;
  for (int k = t; k < 1 + kNeg; k += 256) mx = fmaxf(mx, logits[k]);
  red[t] = mx;
  __syncthreads();
  for (int off = 128; off > 0; off >>= 1) {
    if (t < off) red[t] = fmaxf(red[t], red[t + off]);
    __syncthreads();
  }
  mx = red[0];
  __syncthreads();
  float se = 0.0f;
  for (int k = t; k < 1 + kNeg; k += 256) se += expf(logits[k] - mx);
  red[t] = se;
  __syncthreads();
  for (int off = 128; off > 0; off >>= 1) {
    if (t < off) red[t] += red[t + off];
    __syncthreads();
  }
  if (t == 0 && hardOk[s]) {
    float lse = mx + logf(red[0]);
    atomicAdd(lossAcc, lse - logits[0]);
  }
}

// ---------------- K7: finalize scalar ----------------
__global__ void k7_final(const float* __restrict__ lossAcc, float* __restrict__ out) {
  out[0] = lossAcc[0] / (float)(kQ * kS);
}

// ---------------- launcher ----------------
extern "C" void kernel_launch(void* const* d_in, const int* in_sizes, int n_in,
                              void* d_out, int out_size, void* d_ws, size_t ws_size,
                              hipStream_t stream) {
  const float* rep       = (const float*)d_in[0];
  const float* label     = (const float*)d_in[1];
  const float* mask      = (const float*)d_in[2];
  const float* prob      = (const float*)d_in[3];
  const float* protosIn  = (const float*)d_in[4];
  float* out = (float*)d_out;

  char* ws = (char*)d_ws;
  size_t cur = 0;
  auto alloc = [&](size_t bytes) -> void* {
    void* p = ws + cur;
    cur += (bytes + 255) & ~(size_t)255;
    return p;
  };

  float*    repT           = (float*)alloc((size_t)kN * kC * 4);      // 64 MB
  int*      clsArr         = (int*)alloc((size_t)kN * 4);
  float*    pixNorm        = (float*)alloc((size_t)kN * 4);
  float*    protoSum       = (float*)alloc((size_t)kS * kC * 4);
  unsigned* validCntChunk  = (unsigned*)alloc((size_t)kS * kChunks * 4);
  unsigned* hardCntChunk   = (unsigned*)alloc((size_t)kS * kChunks * 4);
  unsigned* chunkBaseValid = (unsigned*)alloc((size_t)kS * kChunks * 4);
  unsigned* chunkBaseHard  = (unsigned*)alloc((size_t)kS * kChunks * 4);
  unsigned* validTot       = (unsigned*)alloc(kS * 4);
  unsigned* hardTot        = (unsigned*)alloc(kS * 4);
  unsigned* validSegBase   = (unsigned*)alloc(kS * 4);
  unsigned* hardSegBase    = (unsigned*)alloc(kS * 4);
  unsigned* hardOk         = (unsigned*)alloc(kS * 4);
  int*      validList      = (int*)alloc((size_t)kN * 4);
  int*      hardList       = (int*)alloc((size_t)kN * 4);
  float*    proto          = (float*)alloc((size_t)kS * kC * 4);
  float*    protoNorm      = (float*)alloc(kS * 4);
  float*    simCdf         = (float*)alloc((size_t)kS * (kS - 1) * 4);
  int*      anchorIdx      = (int*)alloc((size_t)kS * kQ * 4);
  int*      poolIdx        = (int*)alloc((size_t)kS * kQ * kNeg * 4);  // ~10 MB
  float*    lossAcc        = (float*)alloc(4);
  (void)ws_size; (void)in_sizes; (void)n_in; (void)out_size;

  k0_zero<<<(kS * kC + 255) / 256, 256, 0, stream>>>(protoSum, lossAcc);

  k1_fused<<<kN / kPx, 256, 0, stream>>>(rep, label, mask, prob, repT, clsArr,
                                         pixNorm, protoSum, validCntChunk, hardCntChunk);

  k2_scan<<<1, 64, 0, stream>>>(validCntChunk, hardCntChunk, chunkBaseValid,
                                chunkBaseHard, validTot, hardTot, validSegBase,
                                hardSegBase, hardOk);

  k3_compact<<<kChunks, 32, 0, stream>>>(clsArr, chunkBaseValid, chunkBaseHard,
                                         validSegBase, hardSegBase, validList, hardList);

  k4_proto<<<1, 256, 0, stream>>>(protosIn, protoSum, validTot, proto, protoNorm, simCdf);

  k5_anchors<<<kS, kQ, 0, stream>>>(hardList, hardSegBase, hardTot, anchorIdx);

  k5_pool<<<(kS * kQ * kNeg + 255) / 256, 256, 0, stream>>>(validList, validSegBase,
                                                            validTot, poolIdx);

  dim3 lossGrid(kQ, kS);
  k6_loss<<<lossGrid, 256, 0, stream>>>(repT, pixNorm, proto, protoNorm, anchorIdx,
                                        poolIdx, simCdf, hardOk, lossAcc);

  k7_final<<<1, 1, 0, stream>>>(lossAcc, out);
}
